// GCLSTM_Model_15135464751776
// MI455X (gfx1250) — compile-verified, matrix operand order checked by reading, not run on hardware
//
#include <hip/hip_runtime.h>

typedef __attribute__((ext_vector_type(16))) __bf16 v16bf;
typedef __attribute__((ext_vector_type(8)))  float  v8f;
typedef int v4i __attribute__((vector_size(16)));

#define N_NODES 100000
#define N_EDGES 200000
#define DDIM    256
#define MAT_ELEMS 65536   // 256*256

__device__ inline v8f wmma_bf16(v16bf a, v16bf b, v8f c) {
    return __builtin_amdgcn_wmma_f32_16x16x32_bf16(
        /*neg_a=*/false, a, /*neg_b=*/false, b,
        /*c_mod=*/(short)0, c, /*reuse_a=*/false, /*reuse_b=*/false);
}

__device__ inline float sigm(float x) { return 1.0f / (1.0f + __expf(-x)); }

// Async global -> LDS copy of 16 bytes (one b128 transfer), ASYNCcnt-tracked.
__device__ inline void async_copy_b128(const void* gsrc, void* ldst) {
#if __has_builtin(__builtin_amdgcn_global_load_async_to_lds_b128)
    __builtin_amdgcn_global_load_async_to_lds_b128(
        (__attribute__((address_space(1))) v4i*)(v4i*)const_cast<void*>(gsrc),
        (__attribute__((address_space(3))) v4i*)(v4i*)ldst, 0, 0);
#else
    unsigned lo = (unsigned)(unsigned long long)
        ((__attribute__((address_space(3))) char*)ldst);
    asm volatile("global_load_async_to_lds_b128 %0, %1, off"
                 :: "v"(lo), "v"(gsrc) : "memory");
#endif
}

__device__ inline void wait_async0() {
#if __has_builtin(__builtin_amdgcn_s_wait_asynccnt)
    __builtin_amdgcn_s_wait_asynccnt(0);
#else
    asm volatile("s_wait_asynccnt 0" ::: "memory");
#endif
}

// Load one 16x32 bf16 A-fragment for this lane from a row-major f32 matrix.
// Lane l (l&15 = m) holds row m; half = l>>4 selects K offset +8.
// ISA layout: elems 0..7 -> K = c0..c0+7 ; elems 8..15 -> K = c0+16..c0+23
template<bool RELU>
__device__ inline v16bf load_a_frag(const float* rowBase, int c0) {
    float4 x0 = *reinterpret_cast<const float4*>(rowBase + c0);
    float4 x1 = *reinterpret_cast<const float4*>(rowBase + c0 + 4);
    float4 x2 = *reinterpret_cast<const float4*>(rowBase + c0 + 16);
    float4 x3 = *reinterpret_cast<const float4*>(rowBase + c0 + 20);
    float f[16] = { x0.x, x0.y, x0.z, x0.w, x1.x, x1.y, x1.z, x1.w,
                    x2.x, x2.y, x2.z, x2.w, x3.x, x3.y, x3.z, x3.w };
    union { v16bf v; __bf16 e[16]; } u;
#pragma unroll
    for (int i = 0; i < 16; ++i) {
        float t = RELU ? fmaxf(f[i], 0.0f) : f[i];
        u.e[i] = (__bf16)t;
    }
    return u.v;
}

// ---------------------------------------------------------------------------
// Weight prep: f32 [K=256, N=256] row-major -> bf16 swizzled B-operand layout.
// dst flat index = kc*8192 + nt*512 + lane*16 + j
// k = kc*32 + (j&7) + ((j>>3)<<4) + ((lane>>4)<<3) ; n = nt*16 + (lane&15)
// ---------------------------------------------------------------------------
__global__ __launch_bounds__(256) void prep_weights_kernel(
    const float* __restrict__ Wx, const float* __restrict__ Wh,
    const float* __restrict__ W_lin, const float* __restrict__ W_src,
    const float* __restrict__ W_dst,
    __bf16* __restrict__ swx, __bf16* __restrict__ swh,
    __bf16* __restrict__ swlin, __bf16* __restrict__ swsrc,
    __bf16* __restrict__ swdst)
{
    int tid = blockIdx.x * 256 + threadIdx.x;          // < 11*65536
    int mat = tid >> 16;
    int w   = tid & (MAT_ELEMS - 1);
    int j    = w & 15;
    int lane = (w >> 4) & 31;
    int nt   = (w >> 9) & 15;
    int kc   = w >> 13;
    int k = kc * 32 + (j & 7) + ((j >> 3) << 4) + ((lane >> 4) << 3);
    int n = nt * 16 + (lane & 15);
    int sidx = k * DDIM + n;

    const float* s; __bf16* d;
    if (mat < 4)      { s = Wx    + mat * MAT_ELEMS;       d = swx + mat * MAT_ELEMS; }
    else if (mat < 8) { s = Wh    + (mat - 4) * MAT_ELEMS; d = swh + (mat - 4) * MAT_ELEMS; }
    else if (mat == 8){ s = W_lin;                         d = swlin; }
    else if (mat == 9){ s = W_src;                         d = swsrc; }
    else              { s = W_dst;                         d = swdst; }
    d[w] = (__bf16)s[sidx];
}

// ---------------------------------------------------------------------------
// Kernel A (v2): fused 4-gate GEMM + LSTM cell math.
// Block = 128 threads (4 waves). Block tile = 64 rows x 64 cols:
//   wave w -> row tile (blockIdx.x*4 + w), all waves share cols blockIdx.y*64.
// B fragments for the current k-chunk (Wx+Wh, 4 gates, 4 n-tiles = 32 KB) are
// staged into LDS ONCE per block via async global->LDS b128 copies (ASYNCcnt),
// double-buffered, and consumed via ds_load. 4x less B traffic than per-wave
// global fetch; WMMA math overlaps the next chunk's async staging.
// ---------------------------------------------------------------------------
__global__ __launch_bounds__(128) void gates_kernel(
    const float* __restrict__ nf, const float* __restrict__ h0,
    const float* __restrict__ c0,
    const float* __restrict__ bh, const float* __restrict__ bg,
    const __bf16* __restrict__ swx, const __bf16* __restrict__ swh,
    float* __restrict__ Hout, float* __restrict__ Cout)
{
    __shared__ __align__(16) char ldsB[2][32768];   // [buf][mat][g][nt][lane][32B]

    const int lane = threadIdx.x & 31;
    const int wave = threadIdx.x >> 5;
    const int m    = lane & 15;
    const int half = lane >> 4;
    const int cg4  = blockIdx.y * 4;                 // first n-tile of this block
    const int row_base = (blockIdx.x * 4 + wave) * 16;
    const int rowA = min(row_base + m, N_NODES - 1); // clamp A loads (tail block)
    const size_t arow = (size_t)rowA * DDIM;

    // Stage B fragments for k-chunk kc into buffer buf.
    // 2048 b128 transfers, 16 per thread.
    auto stage = [&](int kc, int buf) {
#pragma unroll
        for (int i = 0; i < 16; ++i) {
            int q     = threadIdx.x * 16 + i;
            int chunk = q & 1;
            int lane2 = (q >> 1) & 31;
            int nt_l  = (q >> 6) & 3;
            int g     = (q >> 8) & 3;
            int matid = (q >> 10) & 1;
            const __bf16* srcm = matid ? swh : swx;
            size_t goff = (size_t)(g * 4096 + (kc * 16 + cg4 + nt_l) * 32 + lane2) * 16
                          + chunk * 8;               // bf16 elements
            int loff = (((matid * 16 + g * 4 + nt_l) * 32 + lane2) * 32) + chunk * 16;
            async_copy_b128(srcm + goff, &ldsB[buf][loff]);
        }
    };

    v8f acc[4][4];
#pragma unroll
    for (int g = 0; g < 4; ++g)
#pragma unroll
        for (int t = 0; t < 4; ++t) acc[g][t] = (v8f)0.0f;

    stage(0, 0);
    for (int kc = 0; kc < 8; ++kc) {
        const int buf = kc & 1;
        wait_async0();        // own async copies for chunk kc done
        __syncthreads();      // -> everyone's copies visible to all waves
        if (kc < 7) stage(kc + 1, buf ^ 1);

        const int c0col = kc * 32 + (half << 3);
        v16bf ax = load_a_frag<false>(nf + arow, c0col);
        v16bf ah = load_a_frag<false>(h0 + arow, c0col);
#pragma unroll
        for (int t = 0; t < 4; ++t) {
#pragma unroll
            for (int g = 0; g < 4; ++g) {
                v16bf bxf = *(const v16bf*)&ldsB[buf][(((0 * 16 + g * 4 + t) * 32 + lane) * 32)];
                acc[g][t] = wmma_bf16(ax, bxf, acc[g][t]);
                v16bf bhf = *(const v16bf*)&ldsB[buf][(((1 * 16 + g * 4 + t) * 32 + lane) * 32)];
                acc[g][t] = wmma_bf16(ah, bhf, acc[g][t]);
            }
        }
    }

    // Epilogue: biases + gate math + C/H stores straight from accumulators.
#pragma unroll
    for (int t = 0; t < 4; ++t) {
        const int col = (cg4 + t) * 16 + m;
        const float bI = bh[0 * DDIM + col] + bg[0 * DDIM + col];
        const float bF = bh[1 * DDIM + col] + bg[1 * DDIM + col];
        const float bT = bh[2 * DDIM + col] + bg[2 * DDIM + col];
        const float bO = bh[3 * DDIM + col] + bg[3 * DDIM + col];
#pragma unroll
        for (int r = 0; r < 8; ++r) {
            const int row = row_base + r + half * 8;
            if (row < N_NODES) {
                const size_t off = (size_t)row * DDIM + col;
                float iv = sigm(acc[0][t][r] + bI);
                float fv = sigm(acc[1][t][r] + bF);
                float tv = tanhf(acc[2][t][r] + bT);
                float ov = sigm(acc[3][t][r] + bO);
                float cn = fv * c0[off] + iv * tv;
                float hn = ov * tanhf(cn);
                Cout[off] = cn;
                Hout[off] = hn;
            }
        }
    }
}

// ---------------------------------------------------------------------------
// Kernel B: z = relu(H) @ W_lin + b_lin   (relu fused into A-fragment load)
// ---------------------------------------------------------------------------
__global__ __launch_bounds__(128) void lin_kernel(
    const float* __restrict__ H, const float* __restrict__ b_lin,
    const v16bf* __restrict__ swlin, float* __restrict__ z)
{
    const int lane = threadIdx.x & 31;
    const int wave = threadIdx.x >> 5;
    const int m    = lane & 15;
    const int half = lane >> 4;
    const int row_base = blockIdx.x * 16;
    const size_t arow = (size_t)(row_base + m) * DDIM;

    v8f acc[4];
#pragma unroll
    for (int t = 0; t < 4; ++t) acc[t] = (v8f)0.0f;

    for (int kc = 0; kc < 8; ++kc) {
        const int c0col = kc * 32 + (half << 3);
        v16bf a = load_a_frag<true>(H + arow, c0col);
#pragma unroll
        for (int t = 0; t < 4; ++t) {
            const int nt = wave * 4 + t;
            v16bf b = swlin[(kc * 16 + nt) * 32 + lane];
            acc[t] = wmma_bf16(a, b, acc[t]);
        }
    }

#pragma unroll
    for (int t = 0; t < 4; ++t) {
        const int col = wave * 64 + t * 16 + m;
        const float bl = b_lin[col];
#pragma unroll
        for (int r = 0; r < 8; ++r) {
            const int row = row_base + r + half * 8;
            z[(size_t)row * DDIM + col] = acc[t][r] + bl;
        }
    }
}

// ---------------------------------------------------------------------------
// Kernel C: edge decoders. 16 edges/block, 4 waves x 64 cols.
// h_src = z[src]@W_src (+b_src); d = z[dst|neg]@W_dst (+b_dst);
// out = sigmoid(relu(h_src + d) @ W_fin + b_fin)
// Final D->1 projection via in-register dot + shfl_xor + LDS ds_add_f32.
// ---------------------------------------------------------------------------
__global__ __launch_bounds__(128) void edge_kernel(
    const float* __restrict__ z,
    const int* __restrict__ src, const int* __restrict__ dst,
    const int* __restrict__ neg,
    const float* __restrict__ b_src, const float* __restrict__ b_dst,
    const float* __restrict__ W_fin, const float* __restrict__ b_fin,
    const v16bf* __restrict__ swsrc, const v16bf* __restrict__ swdst,
    float* __restrict__ posOut, float* __restrict__ negOut)
{
    __shared__ float redP[16];
    __shared__ float redN[16];
    if (threadIdx.x < 16)       redP[threadIdx.x] = 0.0f;
    else if (threadIdx.x < 32)  redN[threadIdx.x - 16] = 0.0f;
    __syncthreads();

    const int lane = threadIdx.x & 31;
    const int wave = threadIdx.x >> 5;
    const int m    = lane & 15;
    const int half = lane >> 4;
    const int e    = blockIdx.x * 16 + m;

    const int si = src[e];
    const int di = dst[e];
    const int ni = neg[e];
    const float* zs = z + (size_t)si * DDIM;
    const float* zd = z + (size_t)di * DDIM;
    const float* zn = z + (size_t)ni * DDIM;

    v8f aS[4], aP[4], aN[4];
#pragma unroll
    for (int t = 0; t < 4; ++t) { aS[t] = (v8f)0.0f; aP[t] = (v8f)0.0f; aN[t] = (v8f)0.0f; }

    for (int kc = 0; kc < 8; ++kc) {
        const int c0col = kc * 32 + (half << 3);
        v16bf fS = load_a_frag<false>(zs, c0col);
        v16bf fD = load_a_frag<false>(zd, c0col);
        v16bf fN = load_a_frag<false>(zn, c0col);
#pragma unroll
        for (int t = 0; t < 4; ++t) {
            const int nt = wave * 4 + t;
            const int fidx = (kc * 16 + nt) * 32 + lane;
            v16bf bS = swsrc[fidx];
            aS[t] = wmma_bf16(fS, bS, aS[t]);
            v16bf bD = swdst[fidx];
            aP[t] = wmma_bf16(fD, bD, aP[t]);
            aN[t] = wmma_bf16(fN, bD, aN[t]);
        }
    }

    // Per-lane partial dot with W_fin across this wave's 64 columns.
    float pp[8], pn[8];
#pragma unroll
    for (int r = 0; r < 8; ++r) { pp[r] = 0.0f; pn[r] = 0.0f; }
#pragma unroll
    for (int t = 0; t < 4; ++t) {
        const int col = wave * 64 + t * 16 + m;
        const float bs = b_src[col];
        const float bd = b_dst[col];
        const float wf = W_fin[col];
#pragma unroll
        for (int r = 0; r < 8; ++r) {
            float hs = aS[t][r] + bs + bd;
            float hp = fmaxf(hs + aP[t][r], 0.0f);
            float hn = fmaxf(hs + aN[t][r], 0.0f);
            pp[r] += hp * wf;
            pn[r] += hn * wf;
        }
    }

    // Reduce across each 16-lane half (rows r / r+8).
#pragma unroll
    for (int off = 8; off >= 1; off >>= 1) {
#pragma unroll
        for (int r = 0; r < 8; ++r) {
            pp[r] += __shfl_xor(pp[r], off, 32);
            pn[r] += __shfl_xor(pn[r], off, 32);
        }
    }
    if (m == 0) {
#pragma unroll
        for (int r = 0; r < 8; ++r) {
            atomicAdd(&redP[r + half * 8], pp[r]);
            atomicAdd(&redN[r + half * 8], pn[r]);
        }
    }
    __syncthreads();

    const float bf = b_fin[0];
    if (threadIdx.x < 16)
        posOut[blockIdx.x * 16 + threadIdx.x] = sigm(redP[threadIdx.x] + bf);
    else if (threadIdx.x < 32)
        negOut[blockIdx.x * 16 + (threadIdx.x - 16)] = sigm(redN[threadIdx.x - 16] + bf);
}

// ---------------------------------------------------------------------------
extern "C" void kernel_launch(void* const* d_in, const int* in_sizes, int n_in,
                              void* d_out, int out_size, void* d_ws, size_t ws_size,
                              hipStream_t stream) {
    const float* node_feat = (const float*)d_in[0];
    const float* h0        = (const float*)d_in[1];
    const float* c0        = (const float*)d_in[2];
    const int*   src       = (const int*)d_in[3];
    const int*   dst       = (const int*)d_in[4];
    const int*   neg       = (const int*)d_in[5];
    const float* Wx        = (const float*)d_in[6];
    const float* Wh        = (const float*)d_in[7];
    const float* bh        = (const float*)d_in[8];
    const float* bg        = (const float*)d_in[9];
    const float* W_lin     = (const float*)d_in[10];
    const float* b_lin     = (const float*)d_in[11];
    const float* W_src     = (const float*)d_in[12];
    const float* b_src     = (const float*)d_in[13];
    const float* W_dst     = (const float*)d_in[14];
    const float* b_dst     = (const float*)d_in[15];
    const float* W_fin     = (const float*)d_in[16];
    const float* b_fin     = (const float*)d_in[17];

    // Workspace carve-up: swizzled bf16 weights, then z buffer.
    char* wsb = (char*)d_ws;
    __bf16* swx   = (__bf16*)wsb;                 // 4 * 65536
    __bf16* swh   = swx   + 4 * MAT_ELEMS;        // 4 * 65536
    __bf16* swlin = swh   + 4 * MAT_ELEMS;        // 65536
    __bf16* swsrc = swlin + MAT_ELEMS;            // 65536
    __bf16* swdst = swsrc + MAT_ELEMS;            // 65536
    float*  z     = (float*)(wsb + 11 * MAT_ELEMS * sizeof(__bf16)); // N*D f32

    float* out    = (float*)d_out;
    float* posOut = out;                                   // [E]
    float* negOut = out + N_EDGES;                         // [E]
    float* Hout   = out + 2 * (size_t)N_EDGES;             // [N,D]
    float* Cout   = Hout + (size_t)N_NODES * DDIM;         // [N,D]

    // 1) One-shot weight conversion + swizzle (11 matrices * 65536 elems).
    prep_weights_kernel<<<(11 * MAT_ELEMS) / 256, 256, 0, stream>>>(
        Wx, Wh, W_lin, W_src, W_dst, swx, swh, swlin, swsrc, swdst);

    // 2) Fused gate GEMMs + LSTM cell math -> H, C.
    //    Grid: 1563 row-blocks of 64 rows (tail clamped) x 4 column groups.
    dim3 ggrid((N_NODES + 63) / 64, 4);
    gates_kernel<<<ggrid, 128, 0, stream>>>(
        node_feat, h0, c0, bh, bg, swx, swh, Hout, Cout);

    // 3) z = relu(H) @ W_lin + b_lin.
    lin_kernel<<<N_NODES / 16, 128, 0, stream>>>(
        Hout, b_lin, (const v16bf*)swlin, z);

    // 4) Edge decoders (pos + neg) with fused W_fin projection.
    edge_kernel<<<N_EDGES / 16, 128, 0, stream>>>(
        z, src, dst, neg, b_src, b_dst, W_fin, b_fin,
        (const v16bf*)swsrc, (const v16bf*)swdst, posOut, negOut);
}
